// DecoderCell_54795192762414
// MI455X (gfx1250) — compile-verified, matrix operand order checked by reading, not run on hardware
//
#include <hip/hip_runtime.h>
#include <math.h>

typedef __attribute__((ext_vector_type(16))) __bf16 v16bf;
typedef __attribute__((ext_vector_type(8)))  __bf16 v8bf;
typedef __attribute__((ext_vector_type(4)))  __bf16 v4bf;
typedef __attribute__((ext_vector_type(8)))  float  v8f;

// ---------------------------------------------------------------------------
// WMMA fragment loader for v_wmma_f32_16x16x32_bf16 (wave32).
// 16-bit A-matrix 16x32 layout (ISA 7.12.2): lane&15 = row M; lane>>4 selects
// K-halves {0-7,16-23} (lo lanes) vs {8-15,24-31} (hi lanes). Each half is a
// CONTIGUOUS run of 8 bf16 -> a fragment is exactly two 16-byte loads.
// B (32x16) mirrors A with lane&15 = column N, so row-major W[n][k] (A @ W^T)
// uses the identical loader. Works for global or LDS pointers (after inlining
// the address space is known, so LDS sources lower to ds_load_b128).
// ---------------------------------------------------------------------------
__device__ inline v16bf frag_bf16(const __bf16* __restrict__ base, int stride, int lane) {
    int r  = lane & 15;
    int kh = (lane >> 4) << 3;                 // 0 or 8
    const __bf16* row = base + (size_t)r * stride + kh;
    v8bf lo = *(const v8bf*)(row);             // k = kh .. kh+7
    v8bf hi = *(const v8bf*)(row + 16);        // k = 16+kh .. 16+kh+7
    return __builtin_shufflevector(lo, hi, 0, 1, 2, 3, 4, 5, 6, 7,
                                           8, 9, 10, 11, 12, 13, 14, 15);
}

__device__ inline float sigf(float x) { return 1.f / (1.f + __expf(-x)); }

// ---------------------------------------------------------------------------
// fp32 -> bf16 conversion (grid-stride)
// ---------------------------------------------------------------------------
__global__ void k_cvt(const float* __restrict__ src, __bf16* __restrict__ dst, int n) {
    int i = blockIdx.x * blockDim.x + threadIdx.x;
    int stride = gridDim.x * blockDim.x;
    for (; i < n; i += stride) dst[i] = (__bf16)src[i];
}

// ---------------------------------------------------------------------------
// ctx[b,c] = sum_t w[b,t] * memory[b,t,c]   (bandwidth-bound, coalesced in c)
// writes into x_dec0[:, 128:640]
// ---------------------------------------------------------------------------
__global__ void __launch_bounds__(256)
k_ctx(const float* __restrict__ w, const float* __restrict__ mem,
      float* __restrict__ xdec0) {
    int b = blockIdx.x;
    int c = threadIdx.x;                       // handles c and c+256
    const float* mb = mem + (size_t)b * 512 * 512;
    const float* wb = w + b * 512;
    float a0 = 0.f, a1 = 0.f;
    for (int t = 0; t < 512; ++t) {
        float wv = wb[t];                      // uniform -> scalar broadcast
        a0 = fmaf(wv, mb[(size_t)t * 512 + c], a0);
        a1 = fmaf(wv, mb[(size_t)t * 512 + c + 256], a1);
    }
    xdec0[b * 640 + 128 + c] = a0;
    xdec0[b * 640 + 384 + c] = a1;
}

// ---------------------------------------------------------------------------
// PreNet (tiny: scalar f32)
// ---------------------------------------------------------------------------
__global__ void __launch_bounds__(128)
k_prenet1(const float* __restrict__ x, const float* __restrict__ W1,
          const float* __restrict__ b1, float* __restrict__ p1) {
    int b = blockIdx.x, u = threadIdx.x;
    const float* xb = x + b * 160;
    const float* wr = W1 + u * 160;
    float acc = b1[u];
    for (int k = 0; k < 160; ++k) acc = fmaf(xb[k], wr[k], acc);
    p1[b * 128 + u] = fmaxf(acc, 0.f);
}

__global__ void __launch_bounds__(128)
k_prenet2(const float* __restrict__ p1, const float* __restrict__ W2,
          const float* __restrict__ b2, float* __restrict__ xdec0) {
    int b = blockIdx.x, u = threadIdx.x;
    const float* xb = p1 + b * 128;
    const float* wr = W2 + u * 128;
    float acc = b2[u];
    for (int k = 0; k < 128; ++k) acc = fmaf(xb[k], wr[k], acc);
    xdec0[b * 640 + u] = fmaxf(acc, 0.f);     // cols 0..127; ctx fills the rest
}

// ---------------------------------------------------------------------------
// gates[m, n] = sum_k x[m,k]*Wih[n,k] + sum_k h[m,k]*Whh[n,k] + bias[n]
// one wave per 16x16 tile; all operands bf16 -> every frag is 2x b128 loads.
// K = 640 + 640 in steps of 32 (40 WMMA / wave). Weights are L2-resident.
// ---------------------------------------------------------------------------
__global__ void __launch_bounds__(32)
k_gates(const __bf16* __restrict__ xin, const __bf16* __restrict__ hin,
        const __bf16* __restrict__ Wih, const __bf16* __restrict__ Whh,
        const float* __restrict__ bias, float* __restrict__ gates) {
    int lane = threadIdx.x & 31;
    int m0 = blockIdx.x * 16;                  // batch tile
    int n0 = blockIdx.y * 16;                  // gate tile
    v8f c = {};
    for (int k = 0; k < 640; k += 32) {
        v16bf a  = frag_bf16(xin + (size_t)m0 * 640 + k, 640, lane);
        v16bf bf = frag_bf16(Wih + (size_t)n0 * 640 + k, 640, lane);
        c = __builtin_amdgcn_wmma_f32_16x16x32_bf16(false, a, false, bf, (short)0, c, false, false);
    }
    for (int k = 0; k < 640; k += 32) {
        v16bf a  = frag_bf16(hin + (size_t)m0 * 640 + k, 640, lane);
        v16bf bf = frag_bf16(Whh + (size_t)n0 * 640 + k, 640, lane);
        c = __builtin_amdgcn_wmma_f32_16x16x32_bf16(false, a, false, bf, (short)0, c, false, false);
    }
    int n = lane & 15, kh = lane >> 4;
    float bv = bias[n0 + n];
#pragma unroll
    for (int e = 0; e < 8; ++e) {
        int row = e + 8 * kh;                  // D layout: VGPR e -> M=e (+8 for hi lanes)
        gates[(size_t)(m0 + row) * 2560 + n0 + n] = c[e] + bv;
    }
}

// ---------------------------------------------------------------------------
// LSTM nonlinearity + residual; also emits bf16 copy of x_out for the next
// WMMA GEMM stage.
// ---------------------------------------------------------------------------
__global__ void __launch_bounds__(256)
k_lstm_elem(const float* __restrict__ gates, const float* __restrict__ cprev,
            const float* __restrict__ xin,
            float* __restrict__ hout, float* __restrict__ cout,
            float* __restrict__ xout, __bf16* __restrict__ xoutbf) {
    int idx = blockIdx.x * blockDim.x + threadIdx.x;   // 0..163839
    int b = idx / 640, u = idx - b * 640;
    const float* g = gates + (size_t)b * 2560;
    float ig = sigf(g[u]);
    float fg = sigf(g[640 + u]);
    float gg = tanhf(g[1280 + u]);
    float og = sigf(g[1920 + u]);
    float c2 = fg * cprev[idx] + ig * gg;
    float h2 = og * tanhf(c2);
    hout[idx] = h2;
    cout[idx] = c2;
    float xo = h2 + xin[idx];
    xout[idx] = xo;
    xoutbf[idx] = (__bf16)xo;
}

// ---------------------------------------------------------------------------
// q = x_dec @ att_fc_W^T + att_fc_b (tiny: scalar f32)
// ---------------------------------------------------------------------------
__global__ void __launch_bounds__(128)
k_q(const float* __restrict__ xdec, const float* __restrict__ W,
    const float* __restrict__ bias, float* __restrict__ q) {
    int b = blockIdx.x, a = threadIdx.x;
    const float* xb = xdec + b * 640;
    const float* wr = W + a * 640;
    float acc = bias[a];
    for (int k = 0; k < 640; ++k) acc = fmaf(xb[k], wr[k], acc);
    q[b * 128 + a] = acc;
}

// ---------------------------------------------------------------------------
// scores[b,t] = sum_a tanh(memory[b,t,:]@keyW[a,:] + keyb[a]) * q[b,a]
// One block per (b, 16-row T-tile). The 16x512 fp32 memory tile is staged
// ONCE into LDS as bf16 (cooperative float4 loads + cvt), removing the 8x
// redundant global reads. Row pad = 8 bf16 so frag ds_load_b128 reads are
// bank-conflict-free (lane r -> banks 4r..4r+3). 8 waves each own a 16-wide
// attention strip: 16 WMMAs over K=512, tanh*q, shuffle-reduce the 16
// N-lanes, LDS-atomic combine across waves. One pass over memory (268 MB).
// ---------------------------------------------------------------------------
#define SA_STRIDE 520   // 512 + 8 bf16 pad
__global__ void __launch_bounds__(256)
k_scores(const float* __restrict__ mem, const __bf16* __restrict__ keyW,
         const float* __restrict__ keyb, const float* __restrict__ q,
         const unsigned char* __restrict__ mmask, float* __restrict__ scores) {
    __shared__ __bf16 sA[16 * SA_STRIDE];
    __shared__ float s_acc[16];
    int b = blockIdx.x;
    int t0 = blockIdx.y * 16;
    int tid = threadIdx.x;
    int wv = tid >> 5;
    int lane = tid & 31;
    if (tid < 16) s_acc[tid] = 0.f;

    // stage 16x512 fp32 -> bf16 LDS tile (8 float4 per thread, coalesced)
    const float4* src = (const float4*)(mem + ((size_t)b * 512 + t0) * 512);
#pragma unroll
    for (int i = 0; i < 8; ++i) {
        int idx = tid + i * 256;               // 0..2047 float4 slots
        int row = idx >> 7;                    // /128
        int c4  = idx & 127;
        float4 f = src[row * 128 + c4];
        v4bf o = { (__bf16)f.x, (__bf16)f.y, (__bf16)f.z, (__bf16)f.w };
        *(v4bf*)(sA + row * SA_STRIDE + c4 * 4) = o;
    }
    __syncthreads();

    int n0 = wv * 16;
    v8f c = {};
    for (int k = 0; k < 512; k += 32) {
        v16bf a  = frag_bf16(sA + k, SA_STRIDE, lane);
        v16bf bf = frag_bf16(keyW + (size_t)n0 * 512 + k, 512, lane);
        c = __builtin_amdgcn_wmma_f32_16x16x32_bf16(false, a, false, bf, (short)0, c, false, false);
    }
    int n = lane & 15, kh = lane >> 4;
    float kb = keyb[n0 + n];
    float qv = q[b * 128 + n0 + n];
#pragma unroll
    for (int e = 0; e < 8; ++e) {
        float v = tanhf(c[e] + kb) * qv;
#pragma unroll
        for (int off = 1; off < 16; off <<= 1) v += __shfl_xor(v, off, 32);
        if ((lane & 15) == 0) atomicAdd(&s_acc[e + 8 * kh], v);
    }
    __syncthreads();
    if (tid < 16) {
        float s = s_acc[tid];
        if (!mmask[b * 512 + t0 + tid]) s = -1e9f;
        scores[b * 512 + t0 + tid] = s;
    }
}

// ---------------------------------------------------------------------------
// softmax over T + Markov prior + renormalize (one block per batch row)
// ---------------------------------------------------------------------------
__global__ void __launch_bounds__(256)
k_softmax_markov(const float* __restrict__ scores, const float* __restrict__ w,
                 float* __restrict__ wnew) {
    __shared__ float red[256];
    int b = blockIdx.x, tid = threadIdx.x;
    const float* sb = scores + b * 512;
    float s0 = sb[tid], s1 = sb[tid + 256];

    red[tid] = fmaxf(s0, s1);
    __syncthreads();
    for (int o = 128; o > 0; o >>= 1) {
        if (tid < o) red[tid] = fmaxf(red[tid], red[tid + o]);
        __syncthreads();
    }
    float m = red[0];
    __syncthreads();

    float e0 = __expf(s0 - m), e1 = __expf(s1 - m);
    red[tid] = e0 + e1;
    __syncthreads();
    for (int o = 128; o > 0; o >>= 1) {
        if (tid < o) red[tid] += red[tid + o];
        __syncthreads();
    }
    float denom = red[0];
    __syncthreads();

    const float* wb = w + b * 512;
    int t0 = tid, t1 = tid + 256;
    float wp0 = 0.5f * wb[t0] + 0.5f * ((t0 > 0) ? wb[t0 - 1] : 0.f);
    float wp1 = 0.5f * wb[t1] + 0.5f * wb[t1 - 1];
    float u0 = wp0 * (e0 / denom);
    float u1 = wp1 * (e1 / denom);

    red[tid] = u0 + u1;
    __syncthreads();
    for (int o = 128; o > 0; o >>= 1) {
        if (tid < o) red[tid] += red[tid + o];
        __syncthreads();
    }
    float tot = red[0] + 1e-8f;
    wnew[b * 512 + t0] = u0 / tot;
    wnew[b * 512 + t1] = u1 / tot;
}

// ---------------------------------------------------------------------------
extern "C" void kernel_launch(void* const* d_in, const int* in_sizes, int n_in,
                              void* d_out, int out_size, void* d_ws, size_t ws_size,
                              hipStream_t stream) {
    (void)in_sizes; (void)n_in; (void)out_size; (void)ws_size;

    const float* x      = (const float*)d_in[0];
    const float* w      = (const float*)d_in[1];
    const float* h0_h   = (const float*)d_in[2];
    const float* h0_c   = (const float*)d_in[3];
    const float* h1_h   = (const float*)d_in[4];
    const float* h1_c   = (const float*)d_in[5];
    const float* memory = (const float*)d_in[6];
    const unsigned char* mmask = (const unsigned char*)d_in[7];
    const float* pre_W1 = (const float*)d_in[8];
    const float* pre_b1 = (const float*)d_in[9];
    const float* pre_W2 = (const float*)d_in[10];
    const float* pre_b2 = (const float*)d_in[11];
    const float* att_W  = (const float*)d_in[12];
    const float* att_b  = (const float*)d_in[13];
    const float* key_W  = (const float*)d_in[14];
    const float* key_b  = (const float*)d_in[15];
    const float* l0_Wih = (const float*)d_in[16];
    const float* l0_Whh = (const float*)d_in[17];
    const float* l0_b   = (const float*)d_in[18];
    const float* l1_Wih = (const float*)d_in[19];
    const float* l1_Whh = (const float*)d_in[20];
    const float* l1_b   = (const float*)d_in[21];

    // output layout: x_dec, w_new, h0, c0, h1, c1 (flat, in return order)
    float* out    = (float*)d_out;
    float* o_xdec = out;                 // 256*640
    float* o_wnew = out + 163840;        // 256*512
    float* o_h0   = out + 294912;
    float* o_c0   = out + 458752;
    float* o_h1   = out + 622592;
    float* o_c1   = out + 786432;

    // workspace carve-up
    char* ws = (char*)d_ws;
    size_t off = 0;
    auto take = [&](size_t nbytes) -> void* {
        void* p = ws + off;
        off += (nbytes + 255) & ~(size_t)255;
        return p;
    };
    __bf16* bW0ih = (__bf16*)take(1638400 * sizeof(__bf16));
    __bf16* bW0hh = (__bf16*)take(1638400 * sizeof(__bf16));
    __bf16* bW1ih = (__bf16*)take(1638400 * sizeof(__bf16));
    __bf16* bW1hh = (__bf16*)take(1638400 * sizeof(__bf16));
    __bf16* bKeyW = (__bf16*)take(65536 * sizeof(__bf16));
    __bf16* bX0   = (__bf16*)take(163840 * sizeof(__bf16));   // bf16(x_dec0)
    __bf16* bH0   = (__bf16*)take(163840 * sizeof(__bf16));   // bf16(h0_h)
    __bf16* bX1   = (__bf16*)take(163840 * sizeof(__bf16));   // bf16(x_dec1)
    __bf16* bH1   = (__bf16*)take(163840 * sizeof(__bf16));   // bf16(h1_h)
    __bf16* bXd   = (__bf16*)take(163840 * sizeof(__bf16));   // dummy bf16 sink
    float* fXdec0 = (float*)take(163840 * sizeof(float));
    float* fXdec1 = (float*)take(163840 * sizeof(float));
    float* fGates = (float*)take(655360 * sizeof(float));
    float* fP1    = (float*)take(32768 * sizeof(float));
    float* fQ     = (float*)take(32768 * sizeof(float));
    float* fScores= (float*)take(131072 * sizeof(float));

    // 1) one-time bf16 weight + hidden-state conversion (~13.7 MB)
    k_cvt<<<512, 256, 0, stream>>>(l0_Wih, bW0ih, 1638400);
    k_cvt<<<512, 256, 0, stream>>>(l0_Whh, bW0hh, 1638400);
    k_cvt<<<512, 256, 0, stream>>>(l1_Wih, bW1ih, 1638400);
    k_cvt<<<512, 256, 0, stream>>>(l1_Whh, bW1hh, 1638400);
    k_cvt<<<64, 256, 0, stream>>>(key_W, bKeyW, 65536);
    k_cvt<<<160, 256, 0, stream>>>(h0_h, bH0, 163840);
    k_cvt<<<160, 256, 0, stream>>>(h1_h, bH1, 163840);

    // 2) ctx pooling (first pass over memory) + PreNet -> x_dec0 (+bf16 copy)
    k_ctx<<<256, 256, 0, stream>>>(w, memory, fXdec0);
    k_prenet1<<<256, 128, 0, stream>>>(x, pre_W1, pre_b1, fP1);
    k_prenet2<<<256, 128, 0, stream>>>(fP1, pre_W2, pre_b2, fXdec0);
    k_cvt<<<160, 256, 0, stream>>>(fXdec0, bX0, 163840);

    // 3) stacked LSTM cells with residual (WMMA gate GEMMs, all-bf16 frags)
    dim3 gGates(16, 160);
    k_gates<<<gGates, 32, 0, stream>>>(bX0, bH0, bW0ih, bW0hh, l0_b, fGates);
    k_lstm_elem<<<640, 256, 0, stream>>>(fGates, h0_c, fXdec0, o_h0, o_c0, fXdec1, bX1);
    k_gates<<<gGates, 32, 0, stream>>>(bX1, bH1, bW1ih, bW1hh, l1_b, fGates);
    k_lstm_elem<<<640, 256, 0, stream>>>(fGates, h1_c, fXdec1, o_h1, o_c1, o_xdec, bXd);

    // 4) attention: q, fused keys-GEMM + tanh + dot (second pass over memory)
    k_q<<<256, 128, 0, stream>>>(o_xdec, att_W, att_b, fQ);
    dim3 gScores(256, 32);
    k_scores<<<gScores, 256, 0, stream>>>(memory, bKeyW, key_b, fQ, mmask, fScores);

    // 5) softmax + Markov prior + renormalize
    k_softmax_markov<<<256, 256, 0, stream>>>(fScores, w, o_wnew);
}
